// IrisAxisVisionEncoderLayer_26018911879675
// MI455X (gfx1250) — matrix operand order repaired and checked by reading, not compiled, not run on hardware
//
#include <hip/hip_runtime.h>
#include <hip/hip_bf16.h>

// ---------------------------------------------------------------------------
// IrisAxisVisionEncoderLayer for MI455X (gfx1250), bf16 WMMA path.
// B=2, C=128, X=Y=Z=48, HEADS=8, e=16. All matmuls via v_wmma_f32_16x16x32_bf16.
// Workspace: xe f32 (113MB) + acc f32 (113MB) + bf16 weights (~0.5MB) ~ 227MB.
// ---------------------------------------------------------------------------

typedef __attribute__((ext_vector_type(16))) __bf16 v16bf;
typedef __attribute__((ext_vector_type(8)))  float  v8f;

#define DEV static __device__ __forceinline__

constexpr int  kC   = 128;        // channels
constexpr int  kT   = 48;         // tokens per attention row
constexpr int  kH   = 8;          // heads
constexpr int  kCS  = 110592;     // channel stride in (B,C,X,Y,Z) = 48^3
constexpr long kBS  = 14155776;   // batch stride = C * 48^3
constexpr int  kNTC = 28311552;   // total elements = 2*128*48^3

DEV __bf16 f2bf(float f) {
  union { float f; unsigned u; } a; a.f = f;
  unsigned r = a.u + 0x7FFFu + ((a.u >> 16) & 1u);   // round-to-nearest-even
  unsigned short h = (unsigned short)(r >> 16);
  union { unsigned short s; __bf16 b; } o; o.s = h;
  return o.b;
}

DEV v8f wmma_bf16(v16bf a, v16bf b, v8f c) {
  // D(f32 16x16) = A(bf16 16x32) * B(bf16 32x16) + C
  return __builtin_amdgcn_wmma_f32_16x16x32_bf16(false, a, false, b,
                                                 (short)0, c, false, false);
}

// A fragment (16x32 bf16) from row-major bf16 [rows x ld], tile origin (m0,k0).
// ISA layout: lane m = lane&15; halves 0..7 -> k = k0+8*(lane>>4)+h,
//             halves 8..15 -> k = k0+8*(lane>>4)+16+h.
DEV v16bf load_a_bf(const __bf16* s, int ld, int m0, int k0, int lane) {
  const __bf16* p = s + (m0 + (lane & 15)) * ld + k0 + ((lane >> 4) << 3);
  v16bf a;
#pragma unroll
  for (int h = 0; h < 8; ++h) { a[h] = p[h]; a[h + 8] = p[h + 16]; }
  return a;
}

// A fragment with K zero-padding (valid k in [0,kmax) relative to k0).
DEV v16bf load_a_bf_kmax(const __bf16* s, int ld, int m0, int k0, int lane, int kmax) {
  int kb = (lane >> 4) << 3;
  const __bf16* p = s + (m0 + (lane & 15)) * ld + k0 + kb;
  v16bf a;
#pragma unroll
  for (int h = 0; h < 8; ++h) {
    a[h]     = (kb + h      < kmax) ? p[h]      : f2bf(0.f);
    a[h + 8] = (kb + 16 + h < kmax) ? p[h + 16] : f2bf(0.f);
  }
  return a;
}

// A fragment converted on the fly from f32 (softmax probabilities), K-padded.
DEV v16bf load_a_f32(const float* s, int ld, int m0, int k0, int lane, int kmax) {
  int kb = k0 + ((lane >> 4) << 3);
  const float* p = s + (m0 + (lane & 15)) * ld;
  v16bf a;
#pragma unroll
  for (int h = 0; h < 8; ++h) {
    int ka = kb + h, kc = kb + 16 + h;
    a[h]     = (ka < kmax) ? f2bf(p[ka]) : f2bf(0.f);
    a[h + 8] = (kc < kmax) ? f2bf(p[kc]) : f2bf(0.f);
  }
  return a;
}

// B fragment (32x16) where B = W^T and W is row-major [n][k] (weights, or K^T
// from the sK tile). Lane = column n; halves 0..7 -> k = 8*(lane>>4)+h, 8..15 -> +16.
DEV v16bf load_b_wt(const __bf16* w, int ld, int n0, int k0, int lane) {
  const __bf16* p = w + (n0 + (lane & 15)) * ld + k0 + ((lane >> 4) << 3);
  v16bf b;
#pragma unroll
  for (int h = 0; h < 8; ++h) { b[h] = p[h]; b[h + 8] = p[h + 16]; }
  return b;
}

DEV v16bf load_b_wt_kmax(const __bf16* w, int ld, int n0, int k0, int lane, int kmax) {
  int kb = (lane >> 4) << 3;
  const __bf16* p = w + (n0 + (lane & 15)) * ld + k0 + kb;
  v16bf b;
#pragma unroll
  for (int h = 0; h < 8; ++h) {
    b[h]     = (kb + h      < kmax) ? p[h]      : f2bf(0.f);
    b[h + 8] = (kb + 16 + h < kmax) ? p[h + 16] : f2bf(0.f);
  }
  return b;
}

// B fragment where B[k][n] = tile[k][n0+n] (V tile: K = token dim, strided).
DEV v16bf load_b_col(const __bf16* s, int ld, int n0, int k0, int lane, int kmax) {
  int n = n0 + (lane & 15);
  int kb = k0 + ((lane >> 4) << 3);
  v16bf b;
#pragma unroll
  for (int h = 0; h < 8; ++h) {
    int k = kb + h, k2 = kb + 16 + h;
    b[h]     = (k < kmax) ? s[k * ld + n] : f2bf(0.f);
    b[h + 8] = (k2 < kmax) ? s[k2 * ld + n] : f2bf(0.f);
  }
  return b;
}

// ---------------------------------------------------------------------------
__global__ void posembed_kernel(const float* __restrict__ x,
                                const float* __restrict__ px,
                                const float* __restrict__ py,
                                const float* __restrict__ pz,
                                float* __restrict__ xe,
                                float* __restrict__ acc) {
  int i = blockIdx.x * 256 + threadIdx.x;
  if (i >= kNTC) return;
  int z = i % 48;
  int q = i / 48;
  int y = q % 48;  q /= 48;
  int xx = q % 48; q /= 48;
  int c = q % kC;
  float v = 2.f * x[i] + px[c * 48 + xx] + py[c * 48 + y] + pz[c * 48 + z];
  xe[i] = v;
  acc[i] = v;   // residual accumulator, attention outputs add into this
}

__global__ void cvt_bf16_kernel(const float* __restrict__ src,
                                __bf16* __restrict__ dst, int n) {
  int i = blockIdx.x * 256 + threadIdx.x;
  if (i < n) dst[i] = f2bf(src[i]);
}

// ---------------------------------------------------------------------------
// One workgroup (8 wave32) per attention row: full fused axis attention.
// smem: sS f32 [8][48][48] | sX,sQ,sK,sV,sO bf16 [48][128]  => 135168 B
// ---------------------------------------------------------------------------
__global__ void attn_axis_kernel(const float* __restrict__ xe,
                                 const __bf16* __restrict__ wq,
                                 const __bf16* __restrict__ wkv,
                                 const __bf16* __restrict__ wout,
                                 const float* __restrict__ bout,
                                 float* __restrict__ acc,
                                 int tokstride, int s1, int s2) {
  extern __shared__ char smem[];
  float*  sS = (float*)smem;                          // [8][48][48] scores/probs
  __bf16* sX = (__bf16*)(smem + kH * kT * kT * 4);
  __bf16* sQ = sX + kT * kC;
  __bf16* sK = sQ + kT * kC;
  __bf16* sV = sK + kT * kC;
  __bf16* sO = sV + kT * kC;

  const int tid  = threadIdx.x;
  const int lane = tid & 31;
  const int wave = tid >> 5;
  const int lhi  = lane >> 4;
  const int llo  = lane & 15;

  const int row = blockIdx.x;                 // 0..4607  (B * 48 * 48)
  const int b   = row / 2304;
  const int rr  = row - b * 2304;
  const int o1  = rr / 48, o2 = rr - o1 * 48;
  const long base = (long)b * kBS + (long)o1 * s1 + (long)o2 * s2;

  __builtin_prefetch(wq, 0, 1);
  __builtin_prefetch(wkv, 0, 1);
  __builtin_prefetch(wout, 0, 1);

  // ---- stage A: load token tile, fp32 -> bf16 ----
  for (int i = tid; i < kT * kC; i += 256) {
    int t = i >> 7, c = i & 127;
    sX[i] = f2bf(xe[base + (long)c * kCS + (long)t * tokstride]);
  }
  __syncthreads();

  // ---- stage B: Q/K/V projections; wave owns output columns [16w,16w+16) ----
  {
    const int n0 = wave * 16;
    v8f dq[3] = {}; v8f dk[3] = {}; v8f dv[3] = {};
#pragma unroll
    for (int kk = 0; kk < kC; kk += 32) {
      v16bf bq = load_b_wt(wq,           kC, n0, kk, lane);
      v16bf bk = load_b_wt(wkv,          kC, n0, kk, lane);  // kv rows 0..127 = K
      v16bf bv = load_b_wt(wkv + kC * kC, kC, n0, kk, lane); // kv rows 128..255 = V
#pragma unroll
      for (int mi = 0; mi < 3; ++mi) {
        v16bf a = load_a_bf(sX, kC, mi * 16, kk, lane);
        dq[mi] = wmma_bf16(a, bq, dq[mi]);
        dk[mi] = wmma_bf16(a, bk, dk[mi]);
        dv[mi] = wmma_bf16(a, bv, dv[mi]);
      }
    }
#pragma unroll
    for (int mi = 0; mi < 3; ++mi)
#pragma unroll
      for (int r = 0; r < 8; ++r) {
        int m = mi * 16 + lhi * 8 + r;
        int n = n0 + llo;
        sQ[m * kC + n] = f2bf(dq[mi][r]);
        sK[m * kC + n] = f2bf(dk[mi][r]);
        sV[m * kC + n] = f2bf(dv[mi][r]);
      }
  }
  __syncthreads();

  // ---- stage C: scores per head (wave == head): S = Q K^T * e^-0.5 ----
  {
    const int co = wave * 16;                    // head channel offset
    float* Sh = sS + wave * kT * kT;
#pragma unroll
    for (int mi = 0; mi < 3; ++mi) {
      v16bf a = load_a_bf_kmax(sQ, kC, mi * 16, co, lane, 16);  // K=16 pad 32
#pragma unroll
      for (int nj = 0; nj < 3; ++nj) {
        v16bf bb = load_b_wt_kmax(sK, kC, nj * 16, co, lane, 16);
        v8f d = {};
        d = wmma_bf16(a, bb, d);
#pragma unroll
        for (int r = 0; r < 8; ++r)
          Sh[(mi * 16 + lhi * 8 + r) * kT + nj * 16 + llo] = d[r] * 0.25f;
      }
    }
  }
  __syncthreads();

  // ---- stage D: row softmax over 8*48 rows of length 48 (fp32) ----
  for (int r2 = tid; r2 < kH * kT; r2 += 256) {
    float* p = sS + r2 * kT;
    float mx = p[0];
    for (int j = 1; j < kT; ++j) mx = fmaxf(mx, p[j]);
    float sum = 0.f;
    for (int j = 0; j < kT; ++j) { float e = __expf(p[j] - mx); p[j] = e; sum += e; }
    float inv = 1.f / sum;
    for (int j = 0; j < kT; ++j) p[j] *= inv;
  }
  __syncthreads();

  // ---- stage E: O = P V per head; K=48 padded to 64 ----
  {
    const int co = wave * 16;
    const float* Sh = sS + wave * kT * kT;
#pragma unroll
    for (int mi = 0; mi < 3; ++mi) {
      v8f d = {};
#pragma unroll
      for (int kk = 0; kk < 64; kk += 32) {
        v16bf a  = load_a_f32(Sh, kT, mi * 16, kk, lane, kT);
        v16bf bb = load_b_col(sV, kC, co, kk, lane, kT);
        d = wmma_bf16(a, bb, d);
      }
#pragma unroll
      for (int r = 0; r < 8; ++r)
        sO[(mi * 16 + lhi * 8 + r) * kC + co + llo] = f2bf(d[r]);
    }
  }
  __syncthreads();

  // ---- stage F: output projection + bias, accumulate into residual ----
  {
    const int n0 = wave * 16;
    v8f d[3] = {};
#pragma unroll
    for (int kk = 0; kk < kC; kk += 32) {
      v16bf bb = load_b_wt(wout, kC, n0, kk, lane);
#pragma unroll
      for (int mi = 0; mi < 3; ++mi) {
        v16bf a = load_a_bf(sO, kC, mi * 16, kk, lane);
        d[mi] = wmma_bf16(a, bb, d[mi]);
      }
    }
    const int n = n0 + llo;
    const float bo = bout[n];
#pragma unroll
    for (int mi = 0; mi < 3; ++mi)
#pragma unroll
      for (int r = 0; r < 8; ++r) {
        int t = mi * 16 + lhi * 8 + r;
        long g = base + (long)n * kCS + (long)t * tokstride;
        acc[g] += d[mi][r] + bo;   // exclusive per workgroup within an axis pass
      }
  }
}

// ---------------------------------------------------------------------------
// LN -> MLP(relu) -> +res -> LN, one workgroup per 48 contiguous tokens.
// smem: sR f32 [48][128] | sA,sB bf16 [48][128] => 49152 B
// ---------------------------------------------------------------------------
__global__ void lnmlp_kernel(const float* __restrict__ acc,
                             const float* __restrict__ gam,
                             const float* __restrict__ bet,
                             const __bf16* __restrict__ w1, const float* __restrict__ b1,
                             const __bf16* __restrict__ w2, const float* __restrict__ b2,
                             float* __restrict__ out) {
  extern __shared__ char smem[];
  float*  sR = (float*)smem;            // residual (post-LN1 x)
  __bf16* sA = (__bf16*)(sR + kT * kC); // ln(x) bf16
  __bf16* sB = sA + kT * kC;            // relu(h1) bf16

  const int tid = threadIdx.x, lane = tid & 31, wave = tid >> 5;
  const int lhi = lane >> 4, llo = lane & 15;
  const int f0 = blockIdx.x * kT;       // 110592 tokens per batch; 48 | 110592
  const int b  = f0 / kCS;
  const int sp = f0 - b * kCS;
  const float* basep = acc + (long)b * kBS + sp;

  for (int i = tid; i < kT * kC; i += 256) {
    int t = i >> 7, c = i & 127;
    sR[i] = basep[(long)c * kCS + t];
  }
  __syncthreads();

  if (tid < kT) {                       // LayerNorm #1, one thread per token
    float* r = sR + tid * kC;
    float m = 0.f;
    for (int c = 0; c < kC; ++c) m += r[c];
    m *= (1.f / kC);
    float v = 0.f;
    for (int c = 0; c < kC; ++c) { float d = r[c] - m; v += d * d; }
    v *= (1.f / kC);
    float inv = rsqrtf(v + 1e-5f);
    for (int c = 0; c < kC; ++c) {
      float ln = (r[c] - m) * inv * gam[c] + bet[c];
      r[c] = ln;
      sA[tid * kC + c] = f2bf(ln);
    }
  }
  __syncthreads();

  { // h = relu(ln(x) @ w1^T + b1)
    const int n0 = wave * 16;
    v8f d[3] = {};
#pragma unroll
    for (int kk = 0; kk < kC; kk += 32) {
      v16bf bb = load_b_wt(w1, kC, n0, kk, lane);
#pragma unroll
      for (int mi = 0; mi < 3; ++mi) {
        v16bf a = load_a_bf(sA, kC, mi * 16, kk, lane);
        d[mi] = wmma_bf16(a, bb, d[mi]);
      }
    }
    const int n = n0 + llo;
    const float bb1 = b1[n];
#pragma unroll
    for (int mi = 0; mi < 3; ++mi)
#pragma unroll
      for (int r = 0; r < 8; ++r)
        sB[(mi * 16 + lhi * 8 + r) * kC + n] = f2bf(fmaxf(d[mi][r] + bb1, 0.f));
  }
  __syncthreads();

  { // y = h @ w2^T + b2 + residual
    const int n0 = wave * 16;
    v8f d[3] = {};
#pragma unroll
    for (int kk = 0; kk < kC; kk += 32) {
      v16bf bb = load_b_wt(w2, kC, n0, kk, lane);
#pragma unroll
      for (int mi = 0; mi < 3; ++mi) {
        v16bf a = load_a_bf(sB, kC, mi * 16, kk, lane);
        d[mi] = wmma_bf16(a, bb, d[mi]);
      }
    }
    const int n = n0 + llo;
    const float bb2 = b2[n];
#pragma unroll
    for (int mi = 0; mi < 3; ++mi)
#pragma unroll
      for (int r = 0; r < 8; ++r)
        sR[(mi * 16 + lhi * 8 + r) * kC + n] += d[mi][r] + bb2;
  }
  __syncthreads();

  if (tid < kT) {                       // LayerNorm #2 + channels-first store
    float* r = sR + tid * kC;
    float m = 0.f;
    for (int c = 0; c < kC; ++c) m += r[c];
    m *= (1.f / kC);
    float v = 0.f;
    for (int c = 0; c < kC; ++c) { float d = r[c] - m; v += d * d; }
    v *= (1.f / kC);
    float inv = rsqrtf(v + 1e-5f);
    float* ob = out + (long)b * kBS + sp + tid;
    for (int c = 0; c < kC; ++c)
      ob[(long)c * kCS] = (r[c] - m) * inv * gam[c] + bet[c];
  }
}

// ---------------------------------------------------------------------------
extern "C" void kernel_launch(void* const* d_in, const int* in_sizes, int n_in,
                              void* d_out, int out_size, void* d_ws, size_t ws_size,
                              hipStream_t stream) {
  (void)in_sizes; (void)n_in; (void)out_size; (void)ws_size;

  const float* x  = (const float*)d_in[0];
  const float* px = (const float*)d_in[1];
  const float* py = (const float*)d_in[2];
  const float* pz = (const float*)d_in[3];
  const float* wq_f[3]   = {(const float*)d_in[4],  (const float*)d_in[8],  (const float*)d_in[12]};
  const float* wkv_f[3]  = {(const float*)d_in[5],  (const float*)d_in[9],  (const float*)d_in[13]};
  const float* wout_f[3] = {(const float*)d_in[6],  (const float*)d_in[10], (const float*)d_in[14]};
  const float* bout_f[3] = {(const float*)d_in[7],  (const float*)d_in[11], (const float*)d_in[15]};
  const float* ln_g = (const float*)d_in[16];
  const float* ln_b = (const float*)d_in[17];
  const float* w1 = (const float*)d_in[18];
  const float* b1 = (const float*)d_in[19];
  const float* w2 = (const float*)d_in[20];
  const float* b2 = (const float*)d_in[21];

  // workspace carve
  float* xe  = (float*)d_ws;
  float* acc = xe + (size_t)kNTC;
  __bf16* p  = (__bf16*)(acc + (size_t)kNTC);
  __bf16 *wq_b[3], *wkv_b[3], *wout_b[3];
  for (int i = 0; i < 3; ++i) {
    wq_b[i]   = p; p += kC * kC;
    wkv_b[i]  = p; p += 2 * kC * kC;
    wout_b[i] = p; p += kC * kC;
  }
  __bf16* w1_b = p; p += kC * kC;
  __bf16* w2_b = p;

  // weights -> bf16
  const int cb = (kC * kC + 255) / 256;
  for (int i = 0; i < 3; ++i) {
    cvt_bf16_kernel<<<cb, 256, 0, stream>>>(wq_f[i], wq_b[i], kC * kC);
    cvt_bf16_kernel<<<2 * cb, 256, 0, stream>>>(wkv_f[i], wkv_b[i], 2 * kC * kC);
    cvt_bf16_kernel<<<cb, 256, 0, stream>>>(wout_f[i], wout_b[i], kC * kC);
  }
  cvt_bf16_kernel<<<cb, 256, 0, stream>>>(w1, w1_b, kC * kC);
  cvt_bf16_kernel<<<cb, 256, 0, stream>>>(w2, w2_b, kC * kC);

  // xe = 2x + pos; acc = xe
  posembed_kernel<<<kNTC / 256, 256, 0, stream>>>(x, px, py, pz, xe, acc);

  // axis attentions (X, Y, Z): (tokstride, s1, s2)
  const size_t shA = (size_t)(kH * kT * kT * 4 + 5 * kT * kC * 2);   // 135168 B
  attn_axis_kernel<<<4608, 256, shA, stream>>>(xe, wq_b[0], wkv_b[0], wout_b[0], bout_f[0], acc, 2304, 48, 1);
  attn_axis_kernel<<<4608, 256, shA, stream>>>(xe, wq_b[1], wkv_b[1], wout_b[1], bout_f[1], acc, 48, 2304, 1);
  attn_axis_kernel<<<4608, 256, shA, stream>>>(xe, wq_b[2], wkv_b[2], wout_b[2], bout_f[2], acc, 1, 2304, 48);

  // LN -> MLP -> LN -> transpose out
  const size_t shM = (size_t)(kT * kC * 4 + 2 * kT * kC * 2);        // 49152 B
  lnmlp_kernel<<<kNTC / kC / kT, 256, shM, stream>>>(acc, ln_g, ln_b, w1_b, b1, w2_b, b2,
                                                     (float*)d_out);
}
